// STGCNKeypoint_6614249636490
// MI455X (gfx1250) — matrix operand order repaired and checked by reading, not compiled, not run on hardware
//
#include <hip/hip_runtime.h>
#include <hip/hip_bf16.h>

typedef __attribute__((ext_vector_type(16))) __bf16 v16bf;
typedef __attribute__((ext_vector_type(8)))  __bf16 v8bf;
typedef __attribute__((ext_vector_type(8)))  float  v8f;

#define BSZ   32
#define TLEN  2048
#define KP    17
#define HIDC  64
#define NCLS  10
#define TT    16              // timesteps per workgroup tile
#define HALO  2               // max dilation
#define TROWS (TT + 2*HALO)   // 20

// ---------------------------------------------------------------------------
// Prep: eff_j[c,o] = sum_i gw[c,i] * tw[o,i,j]  (bf16), plus folded BN params.
// grid = 3 (one per ST-GCN block), block = 256
// ---------------------------------------------------------------------------
__global__ __launch_bounds__(256)
void stgcn_prep(const float* __restrict__ gw0, const float* __restrict__ gw1,
                const float* __restrict__ gw2, const float* __restrict__ tw,
                const float* __restrict__ bns, const float* __restrict__ bnb,
                const float* __restrict__ bnm, const float* __restrict__ bnv,
                __bf16* __restrict__ eff, float* __restrict__ bnp)
{
    const int blk = blockIdx.x;
    const int tid = threadIdx.x;
    const int cin = (blk == 0) ? 3 : HIDC;
    const float* gw = (blk == 0) ? gw0 : ((blk == 1) ? gw1 : gw2);

    for (int idx = tid; idx < HIDC * HIDC; idx += 256) {
        const int o = idx >> 6;
        const int c = idx & 63;
        float s0 = 0.f, s1 = 0.f, s2 = 0.f;
        if (c < cin) {
            for (int i = 0; i < HIDC; ++i) {
                const float g = gw[c * HIDC + i];
                const float* t3 = tw + ((size_t)((blk * HIDC + o) * HIDC + i)) * 3;
                s0 += g * t3[0];
                s1 += g * t3[1];
                s2 += g * t3[2];
            }
        }
        // layout: eff[blk][j][o][c], c contiguous
        __bf16* e = eff + (size_t)blk * 3 * HIDC * HIDC;
        e[(0 * HIDC + o) * HIDC + c] = (__bf16)s0;
        e[(1 * HIDC + o) * HIDC + c] = (__bf16)s1;
        e[(2 * HIDC + o) * HIDC + c] = (__bf16)s2;
    }
    if (tid < HIDC) {
        const int i = blk * HIDC + tid;
        const float inv = bns[i] * rsqrtf(bnv[i] + 1e-5f);
        bnp[blk * 2 * HIDC + tid]        = inv;                    // scale
        bnp[blk * 2 * HIDC + HIDC + tid] = bnb[i] - bnm[i] * inv;  // shift
    }
}

// ---------------------------------------------------------------------------
// Fused ST-GCN block: adjacency mix (VALU, into LDS) + 3-tap channel matmul
// (WMMA bf16) + BN + ReLU + residual (from LDS).
// grid = (TLEN/TT, BSZ), block = 256.  Compile-time: IS_FIRST, RESIDUAL, DIL.
// ---------------------------------------------------------------------------
template <int IS_FIRST, int RESIDUAL, int DIL>
__global__ __launch_bounds__(256)
void stgcn_block(const float* __restrict__ kpts,   // [B,T,K,3] (block 0 input)
                 const __bf16* __restrict__ xin,   // [B,64,T,K] (blocks 1,2 input)
                 const float* __restrict__ adj,    // [17,17]
                 const __bf16* __restrict__ eff,   // [3][64][64] this block's taps
                 const float* __restrict__ bnp,    // [2][64] this block's inv/shift
                 __bf16* __restrict__ yout)        // [B,64,T,K]
{
    __shared__ float  adj_s[KP * KP];
    __shared__ __bf16 u[TROWS][KP][HIDC];   // 20*17*64*2 = 43520 B
    __shared__ __bf16 xres[TT][KP][HIDC];   // 16*17*64*2 = 34816 B (residual stash)

    const int tid = threadIdx.x;
    const int b   = blockIdx.y;
    const int t0  = blockIdx.x * TT;

    for (int i = tid; i < KP * KP; i += 256) adj_s[i] = adj[i];
    __syncthreads();

    // ---- Stage 1: u[c,t,v] = sum_k x[c,t,k] * adj[k,v]  (with temporal halo)
    for (int row = tid; row < HIDC * TROWS; row += 256) {
        const int c  = row / TROWS;
        const int tr = row % TROWS;
        const int gt = t0 + tr - HALO;
        float xv[KP];
        bool valid = (gt >= 0) && (gt < TLEN);
        if (valid) {
            if (IS_FIRST) {
                if (c < 3) {
                    const float* p = kpts + ((size_t)(b * TLEN + gt)) * KP * 3 + c;
#pragma unroll
                    for (int k = 0; k < KP; ++k) xv[k] = p[k * 3];
                } else {
                    valid = false;
                }
            } else {
                const __bf16* p = xin + (((size_t)(b * HIDC + c)) * TLEN + gt) * KP;
#pragma unroll
                for (int k = 0; k < KP; ++k) xv[k] = (float)p[k];
            }
        }
        float acc[KP];
#pragma unroll
        for (int v = 0; v < KP; ++v) acc[v] = 0.f;
        if (valid) {
#pragma unroll
            for (int k = 0; k < KP; ++k) {
                const float xk = xv[k];
#pragma unroll
                for (int v = 0; v < KP; ++v) acc[v] += xk * adj_s[k * KP + v];
            }
            // stash raw x for the residual path (interior rows only)
            if (RESIDUAL && tr >= HALO && tr < HALO + TT) {
#pragma unroll
                for (int k = 0; k < KP; ++k) xres[tr - HALO][k][c] = (__bf16)xv[k];
            }
        }
#pragma unroll
        for (int v = 0; v < KP; ++v) u[tr][v][c] = (__bf16)acc[v];
    }
    __syncthreads();

    // ---- Stage 2: WMMA.  out[o, col] = sum_j eff_j^T @ u(shift_j)
    const int lane  = tid & 31;
    const int w     = tid >> 5;
    const int mtile = w & 3;        // 4 tiles of 16 output channels
    const int hi    = lane >> 4;
    const int nl    = lane & 15;

    // A fragments (eff), ISA 16-bit A layout: lane m=nl; hi half selects K sub-rows.
    // elements 0..7  -> K = kc*32 + hi*8 + {0..7}
    // elements 8..15 -> K = kc*32 + 16 + hi*8 + {0..7}
    v16bf afrag[3][2];
    {
        const int o = mtile * 16 + nl;
#pragma unroll
        for (int j = 0; j < 3; ++j)
#pragma unroll
            for (int kc = 0; kc < 2; ++kc) {
                const __bf16* p0 = eff + ((size_t)(j * HIDC + o)) * HIDC + kc * 32 + hi * 8;
                v8bf lo = *(const v8bf*)p0;
                v8bf hh = *(const v8bf*)(p0 + 16);
                afrag[j][kc] = __builtin_shufflevector(lo, hh,
                    0, 1, 2, 3, 4, 5, 6, 7, 8, 9, 10, 11, 12, 13, 14, 15);
            }
    }

    // Folded BN per C/D VGPR r: o = mtile*16 + hi*8 + r   (ISA C/D layout)
    float bn_inv[8], bn_sh[8];
#pragma unroll
    for (int r = 0; r < 8; ++r) {
        const int o = mtile * 16 + hi * 8 + r;
        bn_inv[r] = bnp[o];
        bn_sh[r]  = bnp[HIDC + o];
    }

    // 17 N-tiles of 16 columns each (16*17 = 272 = TT*KP exactly)
    for (int n = (w >> 2); n < 17; n += 2) {
        const int col = n * 16 + nl;
        const int tl  = col / 17;        // local timestep 0..15
        const int v   = col - tl * 17;   // keypoint 0..16
        v8f acc = {0.f, 0.f, 0.f, 0.f, 0.f, 0.f, 0.f, 0.f};
#pragma unroll
        for (int j = 0; j < 3; ++j) {
            const int trow = tl + HALO + (j - 1) * DIL;
            const __bf16* bp = &u[trow][v][0];
#pragma unroll
            for (int kc = 0; kc < 2; ++kc) {
                // ISA 16-bit B layout: lane = column; elements 0..7 -> K = kc*32 + hi*8 + {0..7}, 8..15 -> K = kc*32 + 16 + hi*8 + {0..7}
                const v8bf blo = *(const v8bf*)(bp + kc * 32 + hi * 8);
                const v8bf bhi = *(const v8bf*)(bp + kc * 32 + hi * 8 + 16);
                const v16bf bfrag = __builtin_shufflevector(blo, bhi,
                    0, 1, 2, 3, 4, 5, 6, 7, 8, 9, 10, 11, 12, 13, 14, 15);
                acc = __builtin_amdgcn_wmma_f32_16x16x32_bf16(
                        false, afrag[j][kc], false, bfrag, (short)0, acc, false, false);
            }
        }
        // ---- Epilogue: BN + ReLU + residual (one ds_load_b128), store bf16
        const int gt = t0 + tl;
        const size_t base = (((size_t)(b * HIDC + mtile * 16 + hi * 8)) * TLEN + gt) * KP + v;
        v8bf rv;
        if (RESIDUAL) rv = *(const v8bf*)&xres[tl][v][mtile * 16 + hi * 8];
#pragma unroll
        for (int r = 0; r < 8; ++r) {
            float val = acc[r] * bn_inv[r] + bn_sh[r];
            val = val > 0.f ? val : 0.f;
            if (RESIDUAL) val += (float)rv[r];
            yout[base + (size_t)r * TLEN * KP] = (__bf16)val;
        }
    }
}

// ---------------------------------------------------------------------------
// Mean pool over (T,K) -> LayerNorm -> FC.  grid = BSZ, block = 256.
// ---------------------------------------------------------------------------
__global__ __launch_bounds__(256)
void stgcn_head(const __bf16* __restrict__ y, const float* __restrict__ ln_s,
                const float* __restrict__ ln_b, const float* __restrict__ fc_w,
                const float* __restrict__ fc_b, float* __restrict__ out)
{
    __shared__ float part[256];
    __shared__ float feat[HIDC];
    __shared__ float stats[2];

    const int b   = blockIdx.x;
    const int tid = threadIdx.x;
    const int o   = tid >> 2;   // channel
    const int q   = tid & 3;    // quarter
    const int NTK = TLEN * KP;  // 34816 (divisible by 32)

    const __bf16* p = y + ((size_t)(b * HIDC + o)) * NTK;
    float s = 0.f;
    for (int i = q * 8; i < NTK; i += 32) {
        const v8bf vv = *(const v8bf*)(p + i);
#pragma unroll
        for (int e = 0; e < 8; ++e) s += (float)vv[e];
    }
    part[tid] = s;
    __syncthreads();
    if (q == 0)
        feat[o] = (part[tid] + part[tid + 1] + part[tid + 2] + part[tid + 3]) * (1.f / NTK);
    __syncthreads();
    if (tid == 0) {
        float m = 0.f;
        for (int i = 0; i < HIDC; ++i) m += feat[i];
        m *= (1.f / HIDC);
        float var = 0.f;
        for (int i = 0; i < HIDC; ++i) { const float d = feat[i] - m; var += d * d; }
        var *= (1.f / HIDC);
        stats[0] = m;
        stats[1] = rsqrtf(var + 1e-5f);
    }
    __syncthreads();
    if (tid < NCLS) {
        const float m = stats[0], rs = stats[1];
        float acc = fc_b[tid];
        for (int i = 0; i < HIDC; ++i) {
            const float fn = (feat[i] - m) * rs * ln_s[i] + ln_b[i];
            acc += fn * fc_w[i * NCLS + tid];
        }
        out[b * NCLS + tid] = acc;
    }
}

// ---------------------------------------------------------------------------
extern "C" void kernel_launch(void* const* d_in, const int* in_sizes, int n_in,
                              void* d_out, int out_size, void* d_ws, size_t ws_size,
                              hipStream_t stream)
{
    const float* kpts = (const float*)d_in[0];
    const float* adj  = (const float*)d_in[1];
    const float* gw0  = (const float*)d_in[2];
    const float* gw1  = (const float*)d_in[3];
    const float* gw2  = (const float*)d_in[4];
    const float* tw   = (const float*)d_in[5];
    const float* bns  = (const float*)d_in[6];
    const float* bnb  = (const float*)d_in[7];
    const float* bnm  = (const float*)d_in[8];
    const float* bnv  = (const float*)d_in[9];
    const float* lns  = (const float*)d_in[10];
    const float* lnb  = (const float*)d_in[11];
    const float* fcw  = (const float*)d_in[12];
    const float* fcb  = (const float*)d_in[13];
    float* out        = (float*)d_out;

    char* ws = (char*)d_ws;
    const size_t bufElems = (size_t)BSZ * HIDC * TLEN * KP;     // 71,303,168
    __bf16* buf0 = (__bf16*)ws;                                 // 136 MB
    __bf16* buf1 = (__bf16*)(ws + bufElems * 2);                // 136 MB
    __bf16* eff  = (__bf16*)(ws + bufElems * 4);                // 3*3*64*64 bf16
    float*  bnp  = (float*)(ws + bufElems * 4 + (size_t)3 * 3 * HIDC * HIDC * 2);

    stgcn_prep<<<3, 256, 0, stream>>>(gw0, gw1, gw2, tw, bns, bnb, bnm, bnv, eff, bnp);

    const dim3 grid(TLEN / TT, BSZ);
    const size_t effBlk = (size_t)3 * HIDC * HIDC;

    // block 0: input = kpts (fp32), dilation 1, no residual
    stgcn_block<1, 0, 1><<<grid, 256, 0, stream>>>(
        kpts, (const __bf16*)nullptr, adj, eff + 0 * effBlk, bnp + 0 * 2 * HIDC, buf0);
    // block 1: input = buf0, dilation 1, residual
    stgcn_block<0, 1, 1><<<grid, 256, 0, stream>>>(
        kpts, buf0, adj, eff + 1 * effBlk, bnp + 1 * 2 * HIDC, buf1);
    // block 2: input = buf1, dilation 2, residual
    stgcn_block<0, 1, 2><<<grid, 256, 0, stream>>>(
        kpts, buf1, adj, eff + 2 * effBlk, bnp + 2 * 2 * HIDC, buf0);

    stgcn_head<<<BSZ, 256, 0, stream>>>(buf0, lns, lnb, fcw, fcb, out);
}
